// xLSTMTeacher_84980222919431
// MI455X (gfx1250) — matrix operand order repaired and checked by reading, not compiled, not run on hardware
//
#include <hip/hip_runtime.h>
#include <cstddef>

// ---------------------------------------------------------------------------
// xLSTM teacher forward, B=2048, HID=512, NH=8, HD=64.
// Zero initial state => R matrices and f-gates drop out; mLSTM reduces to
// per-head scalar dots.  GEMMs run on v_wmma_f32_16x16x32_bf16 (wave32),
// register-blocked 2 M-tiles per wave so each k-step issues 5-6 fragment
// loads for 6-8 WMMAs.  Activations are bf16-packed into the WMMA A-fragment
// layout by the *producer*, so GEMM inner loops are pure b128-load + wmma.
// ---------------------------------------------------------------------------

typedef __bf16 bf16_t;
typedef __attribute__((ext_vector_type(16))) bf16_t        v16bf;
typedef __attribute__((ext_vector_type(8)))  float         v8f;
typedef __attribute__((ext_vector_type(4)))  unsigned int  v4u;

union Frag {
  v16bf v;
  unsigned short us[16];
  v4u u4[2];
};

__device__ __forceinline__ unsigned short f2bf(float f) {
  unsigned int u = __float_as_uint(f);
  u += 0x7FFFu + ((u >> 16) & 1u);     // round-to-nearest-even
  return (unsigned short)(u >> 16);
}

// Fragment layout (both A 16x32 and B 32x16, bf16): stored contiguously as
// ((tile0*kT + kTile)*32 + lane)*16 + e, kT = 16 everywhere here.
// A: lane = (m%16) + 16*half; element e -> k = e + 8*half (e<8) else e+8+8*half.
// B: lane = (n%16) + 16*half; element e -> same k as for A (plus kTile*32).
__device__ __forceinline__ void load_frag(Frag& f, const unsigned short* __restrict__ P,
                                          int tile0, int kt, int lane) {
  const v4u* p = (const v4u*)(P + (((size_t)tile0 * 16 + (size_t)kt) * 32 + (size_t)lane) * 16);
  f.u4[0] = p[0];
  f.u4[1] = p[1];
}

// Inverse A-layout map for a column index kk = k % 32:
//   half' = (kk>>3)&1 ;  e = kk - 8*half' - 8*(kk>>4)
__device__ __forceinline__ void a_slot(int kk, int& hp, int& ep) {
  hp = (kk >> 3) & 1;
  ep = kk - 8 * hp - 8 * (kk >> 4);
}

// Convert fp32 weight [N,K] (row-major, used as B = W^T) into fragment order.
__global__ void pack_w_bf16(const float* __restrict__ W, unsigned short* __restrict__ out,
                            int N, int Kdim) {
  const int idx = blockIdx.x * blockDim.x + threadIdx.x;
  const int total = N * Kdim;
  if (idx >= total) return;
  const int kT    = Kdim >> 5;
  const int e     = idx & 15;
  const int lane  = (idx >> 4) & 31;
  const int tIdx  = idx >> 9;
  const int kTile = tIdx % kT;
  const int nTile = tIdx / kT;
  const int n = nTile * 16 + (lane & 15);
  const int k = kTile * 32 + ((e < 8) ? (e + 8 * (lane >> 4)) : (e + 8 + 8 * (lane >> 4)));
  out[idx] = f2bf(W[(size_t)n * Kdim + k]);
}

// h0[b,j] = V[b]*ip_w[j,0] + t[b]*ip_w[j,1] + ip_b[j], written directly into
// packed A-fragment layout (one thread per packed element).
__global__ void input_proj_packed(const float* __restrict__ V, const float* __restrict__ t,
                                  const float* __restrict__ ipw, const float* __restrict__ ipb,
                                  unsigned short* __restrict__ outP) {
  const int idx = blockIdx.x * blockDim.x + threadIdx.x;   // < 2048*512
  const int e     = idx & 15;
  const int lane  = (idx >> 4) & 31;
  const int tIdx  = idx >> 9;
  const int kTile = tIdx & 15;          // kT = 16
  const int mTile = tIdx >> 4;
  const int m    = mTile * 16 + (lane & 15);
  const int half = lane >> 4;
  const int j    = kTile * 32 + ((e < 8) ? (e + 8 * half) : (e + 8 + 8 * half));
  const float val = V[m] * ipw[2 * j] + t[m] * ipw[2 * j + 1] + ipb[j];
  outP[idx] = f2bf(val);
}

// Fused sLSTM: g = act @ W^T + b for live chunks i(0), g(2), o(3), then
// h = sigmoid(g_o) * tanh(exp(g_i) * tanh(g_g)).
// One wave per 32x16 h-region (2 M-tiles x 3 chunks = 6 accumulators).
__global__ __launch_bounds__(256) void slstm_fused_gemm(
    const unsigned short* __restrict__ aP, const unsigned short* __restrict__ Wp,
    const float* __restrict__ bias, float* __restrict__ hR,
    unsigned short* __restrict__ aPout) {
  const int lane  = threadIdx.x & 31;
  const int wave  = blockIdx.x * (blockDim.x >> 5) + (threadIdx.x >> 5);
  const int mPair = wave >> 5;          // 0..63 -> mTiles 2*mPair, 2*mPair+1
  const int jTile = wave & 31;          // 0..31 (column tile within one chunk)
  const int half  = lane >> 4;
  const int nCol  = jTile * 16 + (lane & 15);
  const int mT0   = mPair * 2;

  v8f z = {};
  v8f acc[2][3] = {{z, z, z}, {z, z, z}};

  for (int kt = 0; kt < 16; ++kt) {
    Frag a0, a1, b0, b2, b3;
    load_frag(a0, aP, mT0,        kt, lane);
    load_frag(a1, aP, mT0 + 1,    kt, lane);
    load_frag(b0, Wp, jTile,      kt, lane);   // cols [0,512)     -> i
    load_frag(b2, Wp, jTile + 64, kt, lane);   // cols [1024,1536) -> g
    load_frag(b3, Wp, jTile + 96, kt, lane);   // cols [1536,2048) -> o
    acc[0][0] = __builtin_amdgcn_wmma_f32_16x16x32_bf16(false, a0.v, false, b0.v, (short)0, acc[0][0], false, false);
    acc[0][1] = __builtin_amdgcn_wmma_f32_16x16x32_bf16(false, a0.v, false, b2.v, (short)0, acc[0][1], false, false);
    acc[0][2] = __builtin_amdgcn_wmma_f32_16x16x32_bf16(false, a0.v, false, b3.v, (short)0, acc[0][2], false, false);
    acc[1][0] = __builtin_amdgcn_wmma_f32_16x16x32_bf16(false, a1.v, false, b0.v, (short)0, acc[1][0], false, false);
    acc[1][1] = __builtin_amdgcn_wmma_f32_16x16x32_bf16(false, a1.v, false, b2.v, (short)0, acc[1][1], false, false);
    acc[1][2] = __builtin_amdgcn_wmma_f32_16x16x32_bf16(false, a1.v, false, b3.v, (short)0, acc[1][2], false, false);
  }

  const float bi = bias[nCol];
  const float bg = bias[1024 + nCol];
  const float bo = bias[1536 + nCol];

  // packed-store coordinates for column nCol (constant across rows)
  const int kk  = nCol & 31;
  const int kT2 = nCol >> 5;
  int hp, ep;
  a_slot(kk, hp, ep);

#pragma unroll
  for (int p = 0; p < 2; ++p) {
    const int mTile = mT0 + p;
#pragma unroll
    for (int r = 0; r < 8; ++r) {
      const int mloc = half * 8 + r;             // row within tile
      const int m = mTile * 16 + mloc;
      const float gi = acc[p][0][r] + bi;
      const float gg = acc[p][1][r] + bg;
      const float go = acc[p][2][r] + bo;
      const float cn = expf(gi) * tanhf(gg);
      const float hv = (1.f / (1.f + expf(-go))) * tanhf(cn);
      hR[(size_t)m * 512 + nCol] = hv;
      const int lanep = mloc + 16 * hp;
      aPout[(((size_t)mTile * 16 + kT2) * 32 + lanep) * 16 + ep] = f2bf(hv);
    }
  }
}

// C[2048,512] = act @ W^T[512,512] + bias.  One wave per 32x64 C region
// (2 M-tiles x 4 N-tiles = 8 accumulators per wave).
__global__ __launch_bounds__(256) void gemm_bias_512(
    const unsigned short* __restrict__ aP, const unsigned short* __restrict__ Wp,
    const float* __restrict__ bias, float* __restrict__ outp) {
  const int lane  = threadIdx.x & 31;
  const int wave  = blockIdx.x * (blockDim.x >> 5) + (threadIdx.x >> 5);
  const int mPair = wave >> 3;          // 0..63 -> mTiles 2*mPair, 2*mPair+1
  const int j4    = wave & 7;           // 0..7  -> nTiles j4*4 .. j4*4+3
  const int half  = lane >> 4;
  const int mT0   = mPair * 2;

  v8f z = {};
  v8f acc[2][4] = {{z, z, z, z}, {z, z, z, z}};

  for (int kt = 0; kt < 16; ++kt) {
    Frag a0, a1;
    load_frag(a0, aP, mT0,     kt, lane);
    load_frag(a1, aP, mT0 + 1, kt, lane);
#pragma unroll
    for (int c = 0; c < 4; ++c) {
      Frag b;
      load_frag(b, Wp, j4 * 4 + c, kt, lane);
      acc[0][c] = __builtin_amdgcn_wmma_f32_16x16x32_bf16(false, a0.v, false, b.v, (short)0, acc[0][c], false, false);
      acc[1][c] = __builtin_amdgcn_wmma_f32_16x16x32_bf16(false, a1.v, false, b.v, (short)0, acc[1][c], false, false);
    }
  }

#pragma unroll
  for (int c = 0; c < 4; ++c) {
    const int n = (j4 * 4 + c) * 16 + (lane & 15);
    const float bb = bias[n];
#pragma unroll
    for (int p = 0; p < 2; ++p) {
#pragma unroll
      for (int r = 0; r < 8; ++r) {
        const int m = (mT0 + p) * 16 + half * 8 + r;
        outp[(size_t)m * 512 + n] = acc[p][c][r] + bb;
      }
    }
  }
}

// Per-row mLSTM head math + LayerNorm.  One wave32 per row; each lane owns
// 16 columns (stride 32).  Writes fp32 rows (outR) and packed bf16 (outP).
__global__ __launch_bounds__(256) void mlstm_rowwise(
    const float* __restrict__ x, const float* __restrict__ q,
    const float* __restrict__ k, const float* __restrict__ v,
    const float* __restrict__ igw, const float* __restrict__ igb,
    const float* __restrict__ lng, const float* __restrict__ lnb,
    float* __restrict__ outR, unsigned short* __restrict__ outP) {
  const int lane = threadIdx.x & 31;
  const int row  = blockIdx.x * (blockDim.x >> 5) + (threadIdx.x >> 5);
  const size_t base = (size_t)row * 512;

  float xv[16], qv[16], kv[16], vv[16];
#pragma unroll
  for (int t = 0; t < 16; ++t) {
    const int c = lane + 32 * t;
    xv[t] = x[base + c];
    qv[t] = q[base + c];
    kv[t] = k[base + c];
    vv[t] = v[base + c];
  }

  // input gates i_h = exp(x . igw[h] + igb[h])
  float ig[8];
#pragma unroll
  for (int h = 0; h < 8; ++h) {
    float p = 0.f;
#pragma unroll
    for (int t = 0; t < 16; ++t) p += xv[t] * igw[h * 512 + lane + 32 * t];
#pragma unroll
    for (int o = 16; o >= 1; o >>= 1) p += __shfl_xor(p, o, 32);
    ig[h] = expf(p + igb[h]);
  }

  // per-head dots: columns lane+32t belong to head t>>1
  float kq[8], vq[8];
#pragma unroll
  for (int h = 0; h < 8; ++h) { kq[h] = 0.f; vq[h] = 0.f; }
#pragma unroll
  for (int t = 0; t < 16; ++t) {
    kq[t >> 1] += kv[t] * qv[t];
    vq[t >> 1] += vv[t] * qv[t];
  }
#pragma unroll
  for (int h = 0; h < 8; ++h) {
#pragma unroll
    for (int o = 16; o >= 1; o >>= 1) {
      kq[h] += __shfl_xor(kq[h], o, 32);
      vq[h] += __shfl_xor(vq[h], o, 32);
    }
  }

  float scale[8];
#pragma unroll
  for (int h = 0; h < 8; ++h)
    scale[h] = ig[h] * kq[h] / (ig[h] * vq[h] + ig[h] + 1e-6f);

  float hv[16];
  float s = 0.f;
#pragma unroll
  for (int t = 0; t < 16; ++t) { hv[t] = scale[t >> 1] * vv[t]; s += hv[t]; }
#pragma unroll
  for (int o = 16; o >= 1; o >>= 1) s += __shfl_xor(s, o, 32);
  const float mu = s * (1.f / 512.f);

  float vs = 0.f;
#pragma unroll
  for (int t = 0; t < 16; ++t) { const float d = hv[t] - mu; vs += d * d; }
#pragma unroll
  for (int o = 16; o >= 1; o >>= 1) vs += __shfl_xor(vs, o, 32);
  const float rstd = rsqrtf(vs * (1.f / 512.f) + 1e-5f);

  // packed-store coordinates: kk = lane for every t
  int hp, ep;
  a_slot(lane, hp, ep);
  const int lanep = (row & 15) + 16 * hp;
  const size_t mT16 = (size_t)(row >> 4) * 16;

#pragma unroll
  for (int t = 0; t < 16; ++t) {
    const int c = lane + 32 * t;
    const float y = (hv[t] - mu) * rstd * lng[c] + lnb[c];
    outR[base + c] = y;
    outP[((mT16 + t) * 32 + lanep) * 16 + ep] = f2bf(y);
  }
}

// out[b] = h_row . op_w + op_b  (one wave per row)
__global__ __launch_bounds__(256) void out_proj_kernel(
    const float* __restrict__ h, const float* __restrict__ opw,
    const float* __restrict__ opb, float* __restrict__ out) {
  const int lane = threadIdx.x & 31;
  const int row  = blockIdx.x * (blockDim.x >> 5) + (threadIdx.x >> 5);
  float p = 0.f;
#pragma unroll
  for (int t = 0; t < 16; ++t) {
    const int c = lane + 32 * t;
    p += h[(size_t)row * 512 + c] * opw[c];
  }
#pragma unroll
  for (int o = 16; o >= 1; o >>= 1) p += __shfl_xor(p, o, 32);
  if (lane == 0) out[row] = p + opb[0];
}

extern "C" void kernel_launch(void* const* d_in, const int* in_sizes, int n_in,
                              void* d_out, int out_size, void* d_ws, size_t ws_size,
                              hipStream_t stream) {
  (void)in_sizes; (void)n_in; (void)out_size; (void)ws_size;

  const float* V     = (const float*)d_in[0];
  const float* tt    = (const float*)d_in[1];
  const float* ip_w  = (const float*)d_in[2];
  const float* ip_b  = (const float*)d_in[3];
  const float* op_w  = (const float*)d_in[4];
  const float* op_b  = (const float*)d_in[5];
  const float* s0_W  = (const float*)d_in[6];
  const float* s0_b  = (const float*)d_in[7];
  // d_in[8] = s0_R (unused: multiplies zero state)
  const float* s2_W  = (const float*)d_in[9];
  const float* s2_b  = (const float*)d_in[10];
  // d_in[11] = s2_R (unused)
  const float* m1_Wq = (const float*)d_in[12];
  const float* m1_Wk = (const float*)d_in[13];
  const float* m1_Wv = (const float*)d_in[14];
  const float* m1_bq = (const float*)d_in[15];
  const float* m1_bk = (const float*)d_in[16];
  const float* m1_bv = (const float*)d_in[17];
  const float* m1_igw = (const float*)d_in[18];
  const float* m1_igb = (const float*)d_in[19];
  // d_in[20]/[21] = m1_fgw/fgb (unused: f multiplies zero state)
  const float* m1_lng = (const float*)d_in[22];
  const float* m1_lnb = (const float*)d_in[23];
  const float* m3_Wq = (const float*)d_in[24];
  const float* m3_Wk = (const float*)d_in[25];
  const float* m3_Wv = (const float*)d_in[26];
  const float* m3_bq = (const float*)d_in[27];
  const float* m3_bk = (const float*)d_in[28];
  const float* m3_bv = (const float*)d_in[29];
  const float* m3_igw = (const float*)d_in[30];
  const float* m3_igb = (const float*)d_in[31];
  // d_in[32]/[33] = m3_fgw/fgb (unused)
  const float* m3_lng = (const float*)d_in[34];
  const float* m3_lnb = (const float*)d_in[35];

  // workspace layout (~27.5 MB)
  char* ws = (char*)d_ws;
  const size_t MB = 1024 * 1024;
  unsigned short* aP0 = (unsigned short*)(ws);                     // 2 MB packed act ping
  unsigned short* aP1 = (unsigned short*)(ws + 2 * MB);            // 2 MB packed act pong
  float* hR1 = (float*)(ws + 4 * MB);                              // 4 MB fp32 rows (sLSTM out)
  float* hR2 = (float*)(ws + 8 * MB);                              // 4 MB fp32 rows (mLSTM out)
  float* qb  = (float*)(ws + 12 * MB);                             // 4 MB
  float* kb  = (float*)(ws + 16 * MB);                             // 4 MB
  float* vb  = (float*)(ws + 20 * MB);                             // 4 MB
  unsigned short* wS = (unsigned short*)(ws + 24 * MB);            // 2 MB (sLSTM W, reused)
  unsigned short* wQ = (unsigned short*)(ws + 26 * MB);            // 0.5 MB
  unsigned short* wK = (unsigned short*)(ws + 26 * MB + 524288);   // 0.5 MB
  unsigned short* wV = (unsigned short*)(ws + 27 * MB);            // 0.5 MB

  const dim3 blk(256);

  // ---- layer 0: input projection + sLSTM ----
  pack_w_bf16<<<4096, blk, 0, stream>>>(s0_W, wS, 2048, 512);
  input_proj_packed<<<4096, blk, 0, stream>>>(V, tt, ip_w, ip_b, aP0);
  slstm_fused_gemm<<<256, blk, 0, stream>>>(aP0, wS, s0_b, hR1, aP1);

  // ---- layer 1: mLSTM ----
  pack_w_bf16<<<1024, blk, 0, stream>>>(m1_Wq, wQ, 512, 512);
  pack_w_bf16<<<1024, blk, 0, stream>>>(m1_Wk, wK, 512, 512);
  pack_w_bf16<<<1024, blk, 0, stream>>>(m1_Wv, wV, 512, 512);
  gemm_bias_512<<<64, blk, 0, stream>>>(aP1, wQ, m1_bq, qb);
  gemm_bias_512<<<64, blk, 0, stream>>>(aP1, wK, m1_bk, kb);
  gemm_bias_512<<<64, blk, 0, stream>>>(aP1, wV, m1_bv, vb);
  mlstm_rowwise<<<256, blk, 0, stream>>>(hR1, qb, kb, vb, m1_igw, m1_igb, m1_lng, m1_lnb, hR2, aP0);

  // ---- layer 2: sLSTM ----
  pack_w_bf16<<<4096, blk, 0, stream>>>(s2_W, wS, 2048, 512);
  slstm_fused_gemm<<<256, blk, 0, stream>>>(aP0, wS, s2_b, hR1, aP1);

  // ---- layer 3: mLSTM ----
  pack_w_bf16<<<1024, blk, 0, stream>>>(m3_Wq, wQ, 512, 512);
  pack_w_bf16<<<1024, blk, 0, stream>>>(m3_Wk, wK, 512, 512);
  pack_w_bf16<<<1024, blk, 0, stream>>>(m3_Wv, wV, 512, 512);
  gemm_bias_512<<<64, blk, 0, stream>>>(aP1, wQ, m3_bq, qb);
  gemm_bias_512<<<64, blk, 0, stream>>>(aP1, wK, m3_bk, kb);
  gemm_bias_512<<<64, blk, 0, stream>>>(aP1, wV, m3_bv, vb);
  mlstm_rowwise<<<256, blk, 0, stream>>>(hR1, qb, kb, vb, m3_igw, m3_igb, m3_lng, m3_lnb, hR2, aP0);

  // ---- output projection ----
  out_proj_kernel<<<256, blk, 0, stream>>>(hR2, op_w, op_b, (float*)d_out);
}
